// SoftXOR_65601330479167
// MI455X (gfx1250) — matrix operand order repaired and checked by reading, not compiled, not run on hardware
//
#include <hip/hip_runtime.h>
#include <hip/hip_bf16.h>

// ---------------------------------------------------------------------------
// SoftXOR on MI455X (gfx1250, wave32):
//  K0: one-time f32->bf16 conversion of input / W_xy (halves GEMM bytes,
//      removes all cvt VALU from the WMMA hot loop)
//  K1: xy = input @ W_xy^T + b_xy     via v_wmma_f32_16x16x32_bf16
//  K2: soft-xor via Walsh-Hadamard:   corr = (ex H) . (ey H) H / 256
//      (H = +-1, exact in bf16, generated from popcount, hoisted to regs)
//      x/y tiles staged into LDS with TENSOR_LOAD_TO_LDS (TDM) + tensorcnt
// ---------------------------------------------------------------------------

#define BATCH   256
#define IN_D    1024
#define NWAL    256         // N (WHT length / out width)
#define CH_N    2304        // CHANS * N
#define COPIES  4

typedef __attribute__((ext_vector_type(16))) __bf16        v16bf;
typedef __attribute__((ext_vector_type(8)))  float         v8f;
typedef __attribute__((ext_vector_type(4)))  unsigned int  v4u;
typedef __attribute__((ext_vector_type(8)))  int           v8i;
typedef __attribute__((ext_vector_type(4)))  int           v4i;

union Frag16 { v16bf v; unsigned int d[8]; };

#if defined(__has_builtin)
#  if __has_builtin(__builtin_amdgcn_tensor_load_to_lds)
#    define HAVE_TDM 1
#  endif
#endif

__device__ __forceinline__ unsigned short f2bf(float f) {
    unsigned int x = __float_as_uint(f);
    unsigned int r = x + 0x7FFFu + ((x >> 16) & 1u);   // RNE truncate
    return (unsigned short)(r >> 16);
}
__device__ __forceinline__ unsigned int pack2bf(float lo, float hi) {
    return (unsigned int)f2bf(lo) | ((unsigned int)f2bf(hi) << 16);
}
// Hadamard entry H[k][n] = (-1)^popcount(k&n) as packed bf16 pair (n0, n0+1)
__device__ __forceinline__ unsigned int hpack(int k, int n0) {
    unsigned int lo = (__popc(k & n0)       & 1) ? 0xBF80u : 0x3F80u;
    unsigned int hi = (__popc(k & (n0 + 1)) & 1) ? 0xBF80u : 0x3F80u;
    return lo | (hi << 16);
}

// ---------------------------------------------------------------------------
// Kernel 0: elementwise f32 -> packed bf16 (pairs)
// ---------------------------------------------------------------------------
__global__ __launch_bounds__(256) void cvt_bf16_kernel(
    const float2* __restrict__ src, unsigned int* __restrict__ dst, int npairs)
{
    int i = blockIdx.x * blockDim.x + threadIdx.x;
    const int stride = gridDim.x * blockDim.x;
    for (; i < npairs; i += stride) {
        const float2 f = src[i];
        dst[i] = pack2bf(f.x, f.y);
    }
}

// ---------------------------------------------------------------------------
// Kernel 1: xy[256,2304] = inA[256,1024](bf16) @ W[2304,1024](bf16)^T + bias
// One 16x16 tile per wave; 8 waves/block; 16 Mtiles x 144 Ntiles.
// ---------------------------------------------------------------------------
__global__ __launch_bounds__(256) void gemm_xy_kernel(
    const unsigned short* __restrict__ Ab, const unsigned short* __restrict__ Wb,
    const float* __restrict__ bias, float* __restrict__ xy)
{
    const int lane  = threadIdx.x & 31;
    const int wid   = blockIdx.x * 8 + (threadIdx.x >> 5);
    const int mtile = wid / 144;            // 0..15
    const int ntile = wid % 144;            // 0..143
    const int hi    = lane >> 4;
    const int M     = mtile * 16 + (lane & 15);
    const unsigned short* arow = Ab + (size_t)M * IN_D;

    v8f acc = {};
    for (int kb = 0; kb < IN_D; kb += 32) {
        Frag16 a, b;
        // A layout (ISA 7.12.2): lane=M; per half, K 0-7 / 16-23 contiguous
        const int ab = kb + hi * 8;
        const uint4 q0 = *(const uint4*)(arow + ab);
        const uint4 q1 = *(const uint4*)(arow + ab + 16);
        a.d[0] = q0.x; a.d[1] = q0.y; a.d[2] = q0.z; a.d[3] = q0.w;
        a.d[4] = q1.x; a.d[5] = q1.y; a.d[6] = q1.z; a.d[7] = q1.w;
        // B layout: lane=N; per half, K 0-7 / 16-23 contiguous
        const unsigned short* wrow = Wb + (size_t)(ntile * 16 + (lane & 15)) * IN_D;
        const uint4 p0 = *(const uint4*)(wrow + ab);
        const uint4 p1 = *(const uint4*)(wrow + ab + 16);
        b.d[0] = p0.x; b.d[1] = p0.y; b.d[2] = p0.z; b.d[3] = p0.w;
        b.d[4] = p1.x; b.d[5] = p1.y; b.d[6] = p1.z; b.d[7] = p1.w;
        acc = __builtin_amdgcn_wmma_f32_16x16x32_bf16(
                  false, a.v, false, b.v, (short)0, acc, false, false);
    }
    // C/D layout: VGPR r -> M = 8*hi + r ; N = lane&15
    const int N  = ntile * 16 + (lane & 15);
    const float bn = bias[N];
#pragma unroll
    for (int r = 0; r < 8; ++r) {
        const int Mo = mtile * 16 + hi * 8 + r;
        xy[(size_t)Mo * CH_N + N] = acc[r] + bn;
    }
}

// ---------------------------------------------------------------------------
// Kernel 2: soft-xor + 1x1 conv. 64 blocks (16 m-rows each), 16 waves.
// ---------------------------------------------------------------------------
__global__ __launch_bounds__(512) void softxor_kernel(
    const float* __restrict__ xy, const float* __restrict__ w_out,
    const float* __restrict__ b_out, float* __restrict__ out)
{
    __shared__ float rawX[16][256];           // TDM-staged x tile (f32)
    __shared__ float rawY[16][256];           // TDM-staged y tile (f32)
    __shared__ unsigned short exA[16][264];   // bf16, padded stride
    __shared__ unsigned short eyA[16][264];
    __shared__ unsigned short prA[16][264];
    __shared__ float mxy[16];

    const int tile = blockIdx.x;            // 0..63 (16 m-rows per tile)
    const int lane = threadIdx.x & 31;
    const int wave = threadIdx.x >> 5;      // 0..15
    const int ntile = wave;

#ifdef HAVE_TDM
    // ---- Stage x/y tiles via Tensor Data Mover (wave 0 only) ---------------
    // 3-D tile: (i=256, c=4 stride 256, b=4 stride 2304), data_size=4B.
    if (wave == 0) {
#pragma unroll
        for (int t = 0; t < 2; ++t) {
            const float* gsrc = xy + (size_t)(tile * 4) * CH_N + t * (COPIES * NWAL);
            const unsigned lds_off =
                (unsigned)(unsigned long long)(t ? (void*)&rawY[0][0] : (void*)&rawX[0][0]);
            const unsigned long long ga = (unsigned long long)gsrc;
            v4u g0; v8i g1; v4i g2, g3;
            g0[0] = 0x1u;                                // count=1
            g0[1] = lds_off;                             // lds_addr
            g0[2] = (unsigned)ga;                        // global_addr lo
            g0[3] = (unsigned)(ga >> 32) | (2u << 30);   // global_addr hi | type=2
            g1[0] = (int)(2u << 16);                     // data_size = 4B
            g1[1] = (int)(256u << 16);                   // tensor_dim0 = 256
            g1[2] = (int)(4u << 16);                     // tensor_dim1 = 4
            g1[3] = (int)(256u << 16);                   // tile_dim0 = 256
            g1[4] = (int)(4u | (4u << 16));              // tile_dim1=4, tile_dim2=4
            g1[5] = 256;                                 // tensor_dim0_stride = 256
            g1[6] = (int)((unsigned)CH_N << 16);         // tensor_dim1_stride = 2304
            g1[7] = 0;
            g2[0] = 4; g2[1] = 0; g2[2] = 0; g2[3] = 0;  // tensor_dim2 = 4
            g3[0] = 0; g3[1] = 0; g3[2] = 0; g3[3] = 0;
#if __has_include(<hip/amd_detail/amd_gfx1250_TDM.h>)
            v8i g4 = {};                                 // therock 6-arg form
            __builtin_amdgcn_tensor_load_to_lds(g0, g1, g2, g3, g4, 0);
#else
            __builtin_amdgcn_tensor_load_to_lds(g0, g1, g2, g3, 0);
#endif
        }
        __builtin_amdgcn_s_wait_tensorcnt(0);
    }
#endif

    // ---- Hoist Hadamard B-fragments for this wave's 16-col tile ------------
    Frag16 hh[8];
#pragma unroll
    for (int t = 0; t < 8; ++t) {
        const int K = ntile * 16 + (lane & 15);
#pragma unroll
        for (int v = 0; v < 8; ++v) hh[t].d[v] = hpack(K, t * 32 + (v >> 2) * 16 + (lane >> 4) * 8 + 2 * (v & 3));
    }

#ifndef HAVE_TDM
    // Fallback: vector-load the tiles into LDS
    {
        const int r = wave;
        const int m = tile * 16 + r, b = m >> 2, c = m & 3;
        const float* xrow = xy + (size_t)b * CH_N + c * NWAL;
        const float* yrow = xy + (size_t)b * CH_N + COPIES * NWAL + c * NWAL;
#pragma unroll
        for (int j = 0; j < 8; ++j) {
            rawX[r][lane + 32 * j] = xrow[lane + 32 * j];
            rawY[r][lane + 32 * j] = yrow[lane + 32 * j];
        }
    }
#endif
    __syncthreads();

    // ---- Phase 0: row max + exp -> bf16 LDS (wave w owns row r = w) --------
    {
        const int r = wave;
        float xv[8], yv[8], mx = -3.4e38f, my = -3.4e38f;
#pragma unroll
        for (int j = 0; j < 8; ++j) {
            xv[j] = rawX[r][lane + 32 * j];
            yv[j] = rawY[r][lane + 32 * j];
            mx = fmaxf(mx, xv[j]);  my = fmaxf(my, yv[j]);
        }
#pragma unroll
        for (int s = 16; s >= 1; s >>= 1) {
            mx = fmaxf(mx, __shfl_xor(mx, s, 32));
            my = fmaxf(my, __shfl_xor(my, s, 32));
        }
#pragma unroll
        for (int j = 0; j < 8; ++j) {
            exA[r][lane + 32 * j] = f2bf(__expf(xv[j] - mx));
            eyA[r][lane + 32 * j] = f2bf(__expf(yv[j] - my));
        }
        if (lane == 0) mxy[r] = mx + my;
    }
    __syncthreads();

    // ---- Phase 1: forward WHT of ex, ey; pointwise product -----------------
    const int mrow = lane & 15, hi = lane >> 4, n_local = lane & 15;
    const unsigned short* exrow = &exA[mrow][0];
    const unsigned short* eyrow = &eyA[mrow][0];
    v8f cx = {}, cy = {};
#pragma unroll
    for (int t = 0; t < 8; ++t) {
        const int ab = t * 32 + hi * 8;     // K 0-7 / 16-23 contiguous per half
        Frag16 ax, ay;
        uint4 q;
        q = *(const uint4*)(exrow + ab);
        ax.d[0] = q.x; ax.d[1] = q.y; ax.d[2] = q.z; ax.d[3] = q.w;
        q = *(const uint4*)(exrow + ab + 16);
        ax.d[4] = q.x; ax.d[5] = q.y; ax.d[6] = q.z; ax.d[7] = q.w;
        q = *(const uint4*)(eyrow + ab);
        ay.d[0] = q.x; ay.d[1] = q.y; ay.d[2] = q.z; ay.d[3] = q.w;
        q = *(const uint4*)(eyrow + ab + 16);
        ay.d[4] = q.x; ay.d[5] = q.y; ay.d[6] = q.z; ay.d[7] = q.w;
        cx = __builtin_amdgcn_wmma_f32_16x16x32_bf16(
                 false, ax.v, false, hh[t].v, (short)0, cx, false, false);
        cy = __builtin_amdgcn_wmma_f32_16x16x32_bf16(
                 false, ay.v, false, hh[t].v, (short)0, cy, false, false);
    }
#pragma unroll
    for (int r = 0; r < 8; ++r)             // C/D layout: VGPR r -> M = 8*hi+r
        prA[8 * hi + r][ntile * 16 + n_local] = f2bf(cx[r] * cy[r]);
    __syncthreads();

    // ---- Phase 2: inverse WHT (same H fragments; 1/256 folded into log) ----
    const unsigned short* prrow = &prA[mrow][0];
    v8f acc = {};
#pragma unroll
    for (int t = 0; t < 8; ++t) {
        const int ab = t * 32 + hi * 8;
        Frag16 a;
        uint4 q;
        q = *(const uint4*)(prrow + ab);
        a.d[0] = q.x; a.d[1] = q.y; a.d[2] = q.z; a.d[3] = q.w;
        q = *(const uint4*)(prrow + ab + 16);
        a.d[4] = q.x; a.d[5] = q.y; a.d[6] = q.z; a.d[7] = q.w;
        acc = __builtin_amdgcn_wmma_f32_16x16x32_bf16(
                  false, a.v, false, hh[t].v, (short)0, acc, false, false);
    }

    // ---- Phase 3: z = mx+my+log(corr/256); channel-weighted sum + skip -----
    const float w0 = w_out[0], w1 = w_out[1], w2 = w_out[2], w3 = w_out[3];
    const float wv[4] = {w0, w1, w2, w3};
    float s_lo = 0.f, s_hi = 0.f;            // local batches 2*hi, 2*hi+1
#pragma unroll
    for (int r = 0; r < 8; ++r) {
        const int Mrow = 8 * hi + r;
        const float corr = fmaxf(acc[r], 1e-30f) * (1.0f / 256.0f);
        const float z = mxy[Mrow] + __logf(corr);
        const float w = wv[Mrow & 3];        // channel c = m % 4
        if (r < 4) s_lo += z * w; else s_hi += z * w;
    }
    const int N    = ntile * 16 + n_local;
    const int b_lo = tile * 4 + 2 * hi;
    const int b_hi = b_lo + 1;
    const float wk = w_out[COPIES], bo = b_out[0];
    const float sk_lo = xy[(size_t)b_lo * CH_N + 2 * COPIES * NWAL + N];
    const float sk_hi = xy[(size_t)b_hi * CH_N + 2 * COPIES * NWAL + N];
    out[b_lo * NWAL + N] = s_lo + sk_lo * wk + bo;
    out[b_hi * NWAL + N] = s_hi + sk_hi * wk + bo;
}

// ---------------------------------------------------------------------------
extern "C" void kernel_launch(void* const* d_in, const int* in_sizes, int n_in,
                              void* d_out, int out_size, void* d_ws, size_t ws_size,
                              hipStream_t stream) {
    const float* input = (const float*)d_in[0];   // [256,1024]
    const float* W_xy  = (const float*)d_in[1];   // [2304,1024]
    const float* b_xy  = (const float*)d_in[2];   // [2304]
    const float* w_out = (const float*)d_in[3];   // [5]
    const float* b_out = (const float*)d_in[4];   // [1]
    float* out = (float*)d_out;                   // [256,256]

    unsigned char* ws = (unsigned char*)d_ws;
    float* xy = (float*)ws;                                        // 2,359,296 B
    unsigned short* inb = (unsigned short*)(ws + 2359296);         //   524,288 B
    unsigned short* wb  = (unsigned short*)(ws + 2359296 + 524288);// 4,718,592 B

    cvt_bf16_kernel<<<512, 256, 0, stream>>>((const float2*)input,
                                             (unsigned int*)inb, BATCH * IN_D / 2);
    cvt_bf16_kernel<<<512, 256, 0, stream>>>((const float2*)W_xy,
                                             (unsigned int*)wb, CH_N * IN_D / 2);
    gemm_xy_kernel<<<288, 256, 0, stream>>>(inb, wb, b_xy, xy);
    softxor_kernel<<<64, 512, 0, stream>>>(xy, w_out, b_out, out);
}
